// GPTDecoderLayer_14224931684965
// MI455X (gfx1250) — matrix operand order repaired and checked by reading, not compiled, not run on hardware
//
#include <hip/hip_runtime.h>
#include <cstdint>
#include <cstddef>

// ---------------- problem constants (match reference) ----------------
#define BB     4
#define SSEQ   2048
#define DDIM   1024
#define HHEADS 16
#define DHEAD  64
#define FDIM   4096
#define MROWS  (BB * SSEQ)   // 8192

typedef __attribute__((ext_vector_type(16))) __bf16 v16bf;
typedef __attribute__((ext_vector_type(8)))  float  v8f;

union Frag { uint4 q[2]; v16bf v; };

#define GLOBAL_AS __attribute__((address_space(1)))
#define LDS_AS    __attribute__((address_space(3)))

#if __has_builtin(__builtin_amdgcn_tensor_load_to_lds)
#define HAS_TDM 1
#endif
#if __has_builtin(__builtin_amdgcn_global_load_async_to_lds_b128)
#define HAS_ASYNC 1
#endif

__device__ __forceinline__ void wait_tensorcnt0() {
#if __has_builtin(__builtin_amdgcn_s_wait_tensorcnt)
  __builtin_amdgcn_s_wait_tensorcnt(0);
#else
  asm volatile("s_wait_tensorcnt 0x0" ::: "memory");
#endif
}

__device__ __forceinline__ void wait_asynccnt0() {
#if __has_builtin(__builtin_amdgcn_s_wait_asynccnt)
  __builtin_amdgcn_s_wait_asynccnt(0);
#else
  asm volatile("s_wait_asynccnt 0x0" ::: "memory");
#endif
}

__device__ __forceinline__ unsigned short f2bf(float f) {
  union { float f; unsigned u; } x; x.f = f;
  unsigned u = x.u;
  u += 0x7fffu + ((u >> 16) & 1u);   // round-to-nearest-even
  return (unsigned short)(u >> 16);
}

__device__ __forceinline__ v8f v8f_zero() {
  v8f z;
  #pragma unroll
  for (int i = 0; i < 8; ++i) z[i] = 0.0f;
  return z;
}

#define NEG_BIG (-3.0e38f)

#ifdef HAS_TDM
// Issue one TDM 2D tile load: tile 128 rows x 32 cols of bf16 from a row-major
// [M,K] tensor (row stride K elements) into LDS at lds_off, with LDS padding of
// 4 DWORDs (8 bf16) after every 16 DWORDs (32 bf16) -> padded pitch of 40.
// D# layout per CDNA5 ISA 8.3/8.4 (group0: count/lds/global/type, group1: dims).
__device__ __forceinline__ void tdm_load_tile_a(const unsigned short* gsrc,
                                                unsigned lds_off, int K, int M) {
  typedef unsigned int v4u __attribute__((ext_vector_type(4)));
  typedef int v4i __attribute__((ext_vector_type(4)));
  typedef int v8i __attribute__((ext_vector_type(8)));
  unsigned long long ga = (unsigned long long)(uintptr_t)gsrc;
  v4u g0;
  g0[0] = 1u;                                                  // count=1, user D#
  g0[1] = lds_off;                                             // lds_addr (bytes)
  g0[2] = (unsigned)(ga & 0xffffffffu);                        // global_addr[31:0]
  g0[3] = (unsigned)((ga >> 32) & 0x1ffffffu) | (2u << 30);    // addr[56:32] | type=2
  v8i g1;
  g1[0] = (int)((1u << 16) |            // data_size = 1 -> 2 bytes
                (1u << 20) |            // pad_enable
                (3u << 22) |            // pad_interval: code 3 -> every 16 DWORDs
                (3u << 25));            // pad_amount:   code 3 -> 4 DWORDs
  g1[1] = (int)(((unsigned)K & 0xffffu) << 16);                            // dim0 lo16
  g1[2] = (int)(((unsigned)K >> 16) | (((unsigned)M & 0xffffu) << 16));    // dim0 hi | dim1 lo
  g1[3] = (int)(((unsigned)M >> 16) | (32u << 16));                        // dim1 hi | tile_dim0=32
  g1[4] = 128;                                                             // tile_dim1=128
  g1[5] = K;                                                               // dim0_stride lo32
  g1[6] = 0;
  g1[7] = 0;
  v4i z4; z4[0] = z4[1] = z4[2] = z4[3] = 0;
#if __clang_major__ >= 23
  v8i z8;
  #pragma unroll
  for (int i = 0; i < 8; ++i) z8[i] = 0;
  __builtin_amdgcn_tensor_load_to_lds(g0, g1, z4, z4, z8, 0);
#else
  __builtin_amdgcn_tensor_load_to_lds(g0, g1, z4, z4, 0);
#endif
}
#endif

// ---------------- weight conversion kernels ----------------
__global__ __launch_bounds__(256) void cvt_f32_bf16(const float* __restrict__ src,
                                                    unsigned short* __restrict__ dst) {
  int i = (blockIdx.x * 256 + threadIdx.x) * 4;
  float4 v = *(const float4*)(src + i);
  ushort4 r;
  r.x = f2bf(v.x); r.y = f2bf(v.y); r.z = f2bf(v.z); r.w = f2bf(v.w);
  *(ushort4*)(dst + i) = r;
}

// [H,D,DH] fp32 -> [D, H*DH] bf16 (so q = h @ W is a plain GEMM)
__global__ __launch_bounds__(256) void cvt_qkv_w(const float* __restrict__ src,
                                                 unsigned short* __restrict__ dst) {
  int i = blockIdx.x * 256 + threadIdx.x;        // i = (h*D + d)*DH + e
  int e = i & (DHEAD - 1);
  int d = (i >> 6) & (DDIM - 1);
  int h = i >> 16;                               // D*DH = 65536
  dst[(size_t)d * DDIM + h * DHEAD + e] = f2bf(src[i]);
}

// ---------------- LayerNorm: fp32 in -> bf16 out, one row per block ----------------
__global__ __launch_bounds__(256) void ln_kernel(const float* __restrict__ x,
                                                 const float* __restrict__ g,
                                                 const float* __restrict__ bta,
                                                 unsigned short* __restrict__ out) {
  int row = blockIdx.x;
  int tid = threadIdx.x;
  int w = tid >> 5, lane = tid & 31;
  size_t base = (size_t)row * DDIM;

  float4 v = ((const float4*)(x + base))[tid];
  float s  = v.x + v.y + v.z + v.w;
  float ss = v.x * v.x + v.y * v.y + v.z * v.z + v.w * v.w;
  #pragma unroll
  for (int off = 16; off >= 1; off >>= 1) {
    s  += __shfl_xor(s,  off, 32);
    ss += __shfl_xor(ss, off, 32);
  }
  __shared__ float shs[8], shss[8];
  if (lane == 0) { shs[w] = s; shss[w] = ss; }
  __syncthreads();
  float tot = 0.f, tots = 0.f;
  #pragma unroll
  for (int i = 0; i < 8; ++i) { tot += shs[i]; tots += shss[i]; }
  float mean = tot * (1.0f / DDIM);
  float var  = tots * (1.0f / DDIM) - mean * mean;
  float rinv = rsqrtf(var + 1e-5f);

  float4 gv = ((const float4*)g)[tid];
  float4 bv = ((const float4*)bta)[tid];
  ushort4 r;
  r.x = f2bf((v.x - mean) * rinv * gv.x + bv.x);
  r.y = f2bf((v.y - mean) * rinv * gv.y + bv.y);
  r.z = f2bf((v.z - mean) * rinv * gv.z + bv.z);
  r.w = f2bf((v.w - mean) * rinv * gv.w + bv.w);
  *(ushort4*)(out + base + tid * 4) = r;
}

// ---------------- WMMA GEMM: C[M,N] = A[M,K]@B[K,N] (+bias)(GELU)(+res) ----------------
// BM=128, BN=128, BK=32; 256 threads = 8 waves; wave w owns rows [w*16, w*16+16),
// all 128 columns (one A-frag feeds 8 WMMAs per K-step -> 64 wmma/block/K-step).
template <bool GELU, bool OUT_BF16>
__global__ __launch_bounds__(256) void gemm_wmma(const unsigned short* __restrict__ A,
                                                 const unsigned short* __restrict__ Bm,
                                                 const float* __restrict__ bias,
                                                 const float* __restrict__ res,
                                                 void* __restrict__ Cout,
                                                 int M, int N, int K) {
  constexpr int LDT = 40;   // padded row pitch (elements); 80B keeps 16B alignment
  __shared__ __align__(16) unsigned short At[128 * LDT];
  __shared__ __align__(16) unsigned short Bt[128 * LDT];   // transposed: Bt[n][k]

  int m0 = blockIdx.x * 128, n0 = blockIdx.y * 128;
  int tid = threadIdx.x;
  int w = tid >> 5, lane = tid & 31;
  int lm = lane & 15, kh = lane >> 4;

  v8f c[8];
  #pragma unroll
  for (int t = 0; t < 8; ++t) c[t] = v8f_zero();

  for (int kk = 0; kk < K; kk += 32) {
    __syncthreads();
    // ---- stage A tile 128x32 (row-major, padded pitch 40) ----
#if defined(HAS_TDM)
    if (w == 0) {
      unsigned lds_off = (unsigned)(unsigned long long)(LDS_AS unsigned short*)At;
      tdm_load_tile_a(A + (size_t)m0 * K + kk, lds_off, K, M);
    }
#elif defined(HAS_ASYNC)
    #pragma unroll
    for (int it = 0; it < 2; ++it) {
      int ch = tid + it * 256;
      int row = ch >> 2, seg = ch & 3;
      __builtin_amdgcn_global_load_async_to_lds_b128(
          (GLOBAL_AS void*)(A + (size_t)(m0 + row) * K + kk + seg * 8),
          (LDS_AS void*)(At + row * LDT + seg * 8), 0, 0);
    }
#else
    #pragma unroll
    for (int it = 0; it < 2; ++it) {
      int ch = tid + it * 256;
      int row = ch >> 2, seg = ch & 3;
      uint4 v = *(const uint4*)(A + (size_t)(m0 + row) * K + kk + seg * 8);
      *(uint4*)(At + row * LDT + seg * 8) = v;
    }
#endif
    // ---- stage B tile 32x128 transposed -> Bt[128][32] ----
    #pragma unroll
    for (int it = 0; it < 2; ++it) {
      int ch = tid + it * 256;
      int k = ch >> 4, nch = ch & 15;
      uint4 v = *(const uint4*)(Bm + (size_t)(kk + k) * N + n0 + nch * 8);
      const unsigned short* sp = (const unsigned short*)&v;
      #pragma unroll
      for (int j = 0; j < 8; ++j) Bt[(nch * 8 + j) * LDT + k] = sp[j];
    }
#if defined(HAS_TDM)
    if (w == 0) wait_tensorcnt0();
#elif defined(HAS_ASYNC)
    wait_asynccnt0();
#endif
    __syncthreads();

    Frag a;   // A-frag: lane holds row lm, K chunks {kh*8..+7, 16+kh*8..+7}
    a.q[0] = *(const uint4*)(At + (w * 16 + lm) * LDT + kh * 8);
    a.q[1] = *(const uint4*)(At + (w * 16 + lm) * LDT + 16 + kh * 8);
    #pragma unroll
    for (int nt = 0; nt < 8; ++nt) {
      Frag b;  // B-frag: lane holds column lm (+nt*16), K chunks {kh*8..+7, 16+kh*8..+7}
      b.q[0] = *(const uint4*)(Bt + (nt * 16 + lm) * LDT + kh * 8);
      b.q[1] = *(const uint4*)(Bt + (nt * 16 + lm) * LDT + 16 + kh * 8);
      c[nt] = __builtin_amdgcn_wmma_f32_16x16x32_bf16(false, a.v, false, b.v,
                                                      (short)0, c[nt], false, false);
    }
  }

  // epilogue: C lane layout: row = r + 8*kh, col = lm
  int row_base = m0 + w * 16 + kh * 8;
  #pragma unroll
  for (int nt = 0; nt < 8; ++nt) {
    int col = n0 + nt * 16 + lm;
    float bv = bias ? bias[col] : 0.0f;
    #pragma unroll
    for (int r = 0; r < 8; ++r) {
      int row = row_base + r;
      float v = c[nt][r] + bv;
      if (GELU) v = 0.5f * v * (1.0f + erff(v * 0.70710678118654752f));
      size_t o = (size_t)row * N + col;
      if (res) v += res[o];
      if (OUT_BF16) ((unsigned short*)Cout)[o] = f2bf(v);
      else          ((float*)Cout)[o] = v;
    }
  }
}

// ---------------- Flash attention (causal), Q/K/V/O as bf16 [M, H*DH] ----------------
// grid: (B*H, S/128); 256 threads = 8 waves; wave w owns queries [qt*128+w*16, +16).
__global__ __launch_bounds__(256) void attn_kernel(const unsigned short* __restrict__ Q,
                                                   const unsigned short* __restrict__ Km,
                                                   const unsigned short* __restrict__ V,
                                                   unsigned short* __restrict__ O) {
  constexpr int LDT = 40;
  __shared__ __align__(16) unsigned short Vt[64 * LDT];       // Vt[d][key]
  __shared__ __align__(16) unsigned short Pw[8][16 * LDT];    // per-wave P tile

  int bh = blockIdx.x;
  int b = bh >> 4, h = bh & 15;
  int qt = blockIdx.y;
  int tid = threadIdx.x;
  int w = tid >> 5, lane = tid & 31;
  int lm = lane & 15, kh = lane >> 4;

  size_t rowBase = (size_t)b * SSEQ;
  int col0 = h * DHEAD;
  int q0 = qt * 128 + w * 16;

  // Q fragments for both 32-wide K-steps of DH=64 (direct from global, row-major)
  Frag qa[2];
  #pragma unroll
  for (int s = 0; s < 2; ++s) {
    const unsigned short* qp = Q + (rowBase + q0 + lm) * DDIM + col0 + s * 32;
    qa[s].q[0] = *(const uint4*)(qp + kh * 8);
    qa[s].q[1] = *(const uint4*)(qp + 16 + kh * 8);
  }

  v8f oc[4];
  #pragma unroll
  for (int t = 0; t < 4; ++t) oc[t] = v8f_zero();
  float mrun[8], lrun[8];
  #pragma unroll
  for (int r = 0; r < 8; ++r) { mrun[r] = NEG_BIG; lrun[r] = 0.0f; }

  int nkb = (qt * 128 + 128) >> 5;   // uniform per block
  for (int kb = 0; kb < nkb; ++kb) {
    int k0 = kb * 32;
    __syncthreads();
    // cooperatively stage V block transposed: Vt[d][key]
    {
      int k = tid >> 3, dch = tid & 7;
      uint4 v = *(const uint4*)(V + (rowBase + k0 + k) * DDIM + col0 + dch * 8);
      const unsigned short* sp = (const unsigned short*)&v;
      #pragma unroll
      for (int j = 0; j < 8; ++j) Vt[(dch * 8 + j) * LDT + k] = sp[j];
    }
    __syncthreads();

    // scores S = Q @ K^T : 2 N-tiles x 2 K-steps; K-frag direct from global
    v8f sc[2];
    sc[0] = v8f_zero(); sc[1] = v8f_zero();
    #pragma unroll
    for (int nt = 0; nt < 2; ++nt) {
      #pragma unroll
      for (int s = 0; s < 2; ++s) {
        Frag kf;  // column = key row of K (contiguous d in memory)
        const unsigned short* kp = Km + (rowBase + k0 + nt * 16 + lm) * DDIM + col0 + s * 32;
        kf.q[0] = *(const uint4*)(kp + kh * 8);
        kf.q[1] = *(const uint4*)(kp + 16 + kh * 8);
        sc[nt] = __builtin_amdgcn_wmma_f32_16x16x32_bf16(false, qa[s].v, false, kf.v,
                                                         (short)0, sc[nt], false, false);
      }
    }

    // online softmax (row = q0 + r + 8*kh, replicated across a 16-lane group)
    #pragma unroll
    for (int r = 0; r < 8; ++r) {
      int q = q0 + r + kh * 8;
      float s0 = (k0 + lm      <= q) ? sc[0][r] * 0.125f : NEG_BIG;  // 1/sqrt(64)
      float s1 = (k0 + 16 + lm <= q) ? sc[1][r] * 0.125f : NEG_BIG;
      float mx = fmaxf(s0, s1);
      #pragma unroll
      for (int off = 1; off <= 8; off <<= 1) mx = fmaxf(mx, __shfl_xor(mx, off, 32));
      float mnew = fmaxf(mrun[r], mx);
      float scale = expf(mrun[r] - mnew);
      float p0 = expf(s0 - mnew);
      float p1 = expf(s1 - mnew);
      float rs = p0 + p1;
      #pragma unroll
      for (int off = 1; off <= 8; off <<= 1) rs += __shfl_xor(rs, off, 32);
      lrun[r] = lrun[r] * scale + rs;
      mrun[r] = mnew;
      #pragma unroll
      for (int nt = 0; nt < 4; ++nt) oc[nt][r] *= scale;
      Pw[w][(r + kh * 8) * LDT + lm]      = f2bf(p0);
      Pw[w][(r + kh * 8) * LDT + 16 + lm] = f2bf(p1);
    }

    // O += P @ V  (A-frag from Pw, B-frag from Vt; single K-step of 32 keys)
    Frag pa;
    pa.q[0] = *(const uint4*)(&Pw[w][lm * LDT + kh * 8]);
    pa.q[1] = *(const uint4*)(&Pw[w][lm * LDT + 16 + kh * 8]);
    #pragma unroll
    for (int nt = 0; nt < 4; ++nt) {
      Frag vf;
      vf.q[0] = *(const uint4*)(Vt + (nt * 16 + lm) * LDT + kh * 8);
      vf.q[1] = *(const uint4*)(Vt + (nt * 16 + lm) * LDT + 16 + kh * 8);
      oc[nt] = __builtin_amdgcn_wmma_f32_16x16x32_bf16(false, pa.v, false, vf.v,
                                                       (short)0, oc[nt], false, false);
    }
  }

  // write O (concat layout [B*S, H*DH], bf16)
  #pragma unroll
  for (int r = 0; r < 8; ++r) {
    float inv = 1.0f / lrun[r];
    size_t orow = (rowBase + q0 + r + kh * 8) * DDIM + col0;
    #pragma unroll
    for (int nt = 0; nt < 4; ++nt)
      O[orow + nt * 16 + lm] = f2bf(oc[nt][r] * inv);
  }
}

// ---------------- orchestration ----------------
extern "C" void kernel_launch(void* const* d_in, const int* in_sizes, int n_in,
                              void* d_out, int out_size, void* d_ws, size_t ws_size,
                              hipStream_t stream) {
  (void)in_sizes; (void)n_in; (void)out_size; (void)ws_size;

  const float* x    = (const float*)d_in[0];
  /* d_in[1] = mask (bool) — causal mask computed analytically */
  const float* wq   = (const float*)d_in[2];
  const float* wk   = (const float*)d_in[3];
  const float* wv   = (const float*)d_in[4];
  const float* wo   = (const float*)d_in[5];
  const float* w1   = (const float*)d_in[6];
  const float* b1   = (const float*)d_in[7];
  const float* w2   = (const float*)d_in[8];
  const float* b2   = (const float*)d_in[9];
  const float* ln1g = (const float*)d_in[10];
  const float* ln1b = (const float*)d_in[11];
  const float* ln2g = (const float*)d_in[12];
  const float* ln2b = (const float*)d_in[13];
  float* out = (float*)d_out;
  char* ws = (char*)d_ws;

  // workspace layout (all sizes are 256-byte multiples; Q|K|V|Cc contiguous)
  size_t off = 0;
  auto take = [&](size_t bytes) { size_t r = off; off += (bytes + 255) & ~(size_t)255; return r; };
  unsigned short* h1 = (unsigned short*)(ws + take((size_t)MROWS * DDIM * 2)); // 16 MB
  unsigned short* Wq = (unsigned short*)(ws + take((size_t)DDIM * DDIM * 2));
  unsigned short* Wk = (unsigned short*)(ws + take((size_t)DDIM * DDIM * 2));
  unsigned short* Wv = (unsigned short*)(ws + take((size_t)DDIM * DDIM * 2));
  unsigned short* Wo = (unsigned short*)(ws + take((size_t)DDIM * DDIM * 2));
  unsigned short* W1 = (unsigned short*)(ws + take((size_t)DDIM * FDIM * 2));
  unsigned short* W2 = (unsigned short*)(ws + take((size_t)FDIM * DDIM * 2));
  unsigned short* Qb = (unsigned short*)(ws + take((size_t)MROWS * DDIM * 2));
  unsigned short* Kb = (unsigned short*)(ws + take((size_t)MROWS * DDIM * 2));
  unsigned short* Vb = (unsigned short*)(ws + take((size_t)MROWS * DDIM * 2));
  unsigned short* Cc = (unsigned short*)(ws + take((size_t)MROWS * DDIM * 2));
  float*          X2 = (float*)(ws + take((size_t)MROWS * DDIM * 4));          // 32 MB
  unsigned short* H2 = h1;   // alias: h1 dead after QKV projections
  unsigned short* FFb = Qb;  // alias: Q|K|V|Cc (64 MB contiguous) dead after out-proj

  dim3 blk(256);

  // 1) weight conversions (deterministic each call)
  cvt_qkv_w<<<(DDIM * DDIM) / 256, blk, 0, stream>>>(wq, Wq);
  cvt_qkv_w<<<(DDIM * DDIM) / 256, blk, 0, stream>>>(wk, Wk);
  cvt_qkv_w<<<(DDIM * DDIM) / 256, blk, 0, stream>>>(wv, Wv);
  cvt_f32_bf16<<<(DDIM * DDIM) / 1024, blk, 0, stream>>>(wo, Wo);
  cvt_f32_bf16<<<(DDIM * FDIM) / 1024, blk, 0, stream>>>(w1, W1);
  cvt_f32_bf16<<<(FDIM * DDIM) / 1024, blk, 0, stream>>>(w2, W2);

  // 2) LN1
  ln_kernel<<<MROWS, blk, 0, stream>>>(x, ln1g, ln1b, h1);

  // 3) QKV projections
  dim3 gD(MROWS / 128, DDIM / 128);
  gemm_wmma<false, true><<<gD, blk, 0, stream>>>(h1, Wq, nullptr, nullptr, Qb, MROWS, DDIM, DDIM);
  gemm_wmma<false, true><<<gD, blk, 0, stream>>>(h1, Wk, nullptr, nullptr, Kb, MROWS, DDIM, DDIM);
  gemm_wmma<false, true><<<gD, blk, 0, stream>>>(h1, Wv, nullptr, nullptr, Vb, MROWS, DDIM, DDIM);

  // 4) causal flash attention -> concat
  attn_kernel<<<dim3(BB * HHEADS, SSEQ / 128), blk, 0, stream>>>(Qb, Kb, Vb, Cc);

  // 5) out-projection + residual (fp32)
  gemm_wmma<false, false><<<gD, blk, 0, stream>>>(Cc, Wo, nullptr, x, X2, MROWS, DDIM, DDIM);

  // 6) LN2
  ln_kernel<<<MROWS, blk, 0, stream>>>(X2, ln2g, ln2b, H2);

  // 7) FFN up + bias + exact GELU (bf16)
  dim3 gF(MROWS / 128, FDIM / 128);
  gemm_wmma<true, true><<<gF, blk, 0, stream>>>(H2, W1, b1, nullptr, FFb, MROWS, FDIM, DDIM);

  // 8) FFN down + bias + residual -> fp32 output
  gemm_wmma<false, false><<<gD, blk, 0, stream>>>(FFb, W2, b2, X2, out, MROWS, DDIM, FDIM);
}
